// OptimizedRWKVBlock_1632087572712
// MI455X (gfx1250) — matrix operand order repaired and checked by reading, not compiled, not run on hardware
//
#include <hip/hip_runtime.h>
#include <hip/hip_fp16.h>

typedef __attribute__((ext_vector_type(16))) _Float16 v16h;
typedef __attribute__((ext_vector_type(8)))  _Float16 v8h;
typedef __attribute__((ext_vector_type(4)))  _Float16 h4;
typedef __attribute__((ext_vector_type(8)))  float    v8f;

constexpr int Bsz  = 8;
constexpr int Tlen = 4096;
constexpr int Cdim = 1024;
constexpr int Hdim = 4096;
constexpr int BT   = Bsz * Tlen;          // 32768 tokens

// ---------------------------------------------------------------------------
// fp32 -> fp16 convert (weights), grid-stride
// ---------------------------------------------------------------------------
__global__ __launch_bounds__(256)
void cvt_f32_f16(const float* __restrict__ in, _Float16* __restrict__ out, int n) {
    int i = blockIdx.x * blockDim.x + threadIdx.x;
    int stride = gridDim.x * blockDim.x;
    for (; i < n; i += stride) out[i] = (_Float16)in[i];
}

// ---------------------------------------------------------------------------
// LayerNorm over C=1024, one block (256 threads, float4 per thread) per token
// ---------------------------------------------------------------------------
__global__ __launch_bounds__(256)
void ln1024(const float* __restrict__ x, const float* __restrict__ g,
            const float* __restrict__ b, float* __restrict__ out) {
    __shared__ float sS[256];
    __shared__ float sQ[256];
    const int tid = threadIdx.x;
    const size_t base = (size_t)blockIdx.x * Cdim;
    float4 xv = ((const float4*)(x + base))[tid];
    float s = xv.x + xv.y + xv.z + xv.w;
    float q = xv.x * xv.x + xv.y * xv.y + xv.z * xv.z + xv.w * xv.w;
    sS[tid] = s; sQ[tid] = q;
    __syncthreads();
    for (int off = 128; off > 0; off >>= 1) {
        if (tid < off) { sS[tid] += sS[tid + off]; sQ[tid] += sQ[tid + off]; }
        __syncthreads();
    }
    const float mu   = sS[0] * (1.0f / Cdim);
    const float var  = sQ[0] * (1.0f / Cdim) - mu * mu;
    const float rstd = rsqrtf(var + 1e-5f);
    float4 gg = ((const float4*)g)[tid];
    float4 bb = ((const float4*)b)[tid];
    float4 o;
    o.x = (xv.x - mu) * rstd * gg.x + bb.x;
    o.y = (xv.y - mu) * rstd * gg.y + bb.y;
    o.z = (xv.z - mu) * rstd * gg.z + bb.z;
    o.w = (xv.w - mu) * rstd * gg.w + bb.w;
    ((float4*)(out + base))[tid] = o;
}

// ---------------------------------------------------------------------------
// token-shift mix, 3 outputs (time-mix) -> f16
// ---------------------------------------------------------------------------
__global__ __launch_bounds__(256)
void mix_tm(const float* __restrict__ xa,
            const float* __restrict__ rk, const float* __restrict__ rv,
            const float* __restrict__ rr,
            _Float16* __restrict__ xk, _Float16* __restrict__ xv,
            _Float16* __restrict__ xr) {
    const int n4 = BT * Cdim / 4;
    int i = blockIdx.x * blockDim.x + threadIdx.x;
    const int stride = gridDim.x * blockDim.x;
    const float4* xa4 = (const float4*)xa;
    for (; i < n4; i += stride) {
        const int c4  = i & (Cdim / 4 - 1);
        const int tok = i >> 8;                 // i / (C/4)
        float4 a = xa4[i];
        float4 p = make_float4(0.f, 0.f, 0.f, 0.f);
        if ((tok % Tlen) != 0) p = xa4[i - Cdim / 4];
        float4 K = ((const float4*)rk)[c4];
        float4 V = ((const float4*)rv)[c4];
        float4 R = ((const float4*)rr)[c4];
        h4 ok, ov, orr;
        ok.x = (_Float16)(a.x * K.x + p.x * (1.f - K.x));
        ok.y = (_Float16)(a.y * K.y + p.y * (1.f - K.y));
        ok.z = (_Float16)(a.z * K.z + p.z * (1.f - K.z));
        ok.w = (_Float16)(a.w * K.w + p.w * (1.f - K.w));
        ov.x = (_Float16)(a.x * V.x + p.x * (1.f - V.x));
        ov.y = (_Float16)(a.y * V.y + p.y * (1.f - V.y));
        ov.z = (_Float16)(a.z * V.z + p.z * (1.f - V.z));
        ov.w = (_Float16)(a.w * V.w + p.w * (1.f - V.w));
        orr.x = (_Float16)(a.x * R.x + p.x * (1.f - R.x));
        orr.y = (_Float16)(a.y * R.y + p.y * (1.f - R.y));
        orr.z = (_Float16)(a.z * R.z + p.z * (1.f - R.z));
        orr.w = (_Float16)(a.w * R.w + p.w * (1.f - R.w));
        ((h4*)xk)[i] = ok;
        ((h4*)xv)[i] = ov;
        ((h4*)xr)[i] = orr;
    }
}

// token-shift mix, 2 outputs (channel-mix) -> f16
__global__ __launch_bounds__(256)
void mix_cm(const float* __restrict__ xb,
            const float* __restrict__ crk, const float* __restrict__ crr,
            _Float16* __restrict__ xck, _Float16* __restrict__ xcr) {
    const int n4 = BT * Cdim / 4;
    int i = blockIdx.x * blockDim.x + threadIdx.x;
    const int stride = gridDim.x * blockDim.x;
    const float4* xb4 = (const float4*)xb;
    for (; i < n4; i += stride) {
        const int c4  = i & (Cdim / 4 - 1);
        const int tok = i >> 8;
        float4 a = xb4[i];
        float4 p = make_float4(0.f, 0.f, 0.f, 0.f);
        if ((tok % Tlen) != 0) p = xb4[i - Cdim / 4];
        float4 K = ((const float4*)crk)[c4];
        float4 R = ((const float4*)crr)[c4];
        h4 ok, orr;
        ok.x = (_Float16)(a.x * K.x + p.x * (1.f - K.x));
        ok.y = (_Float16)(a.y * K.y + p.y * (1.f - K.y));
        ok.z = (_Float16)(a.z * K.z + p.z * (1.f - K.z));
        ok.w = (_Float16)(a.w * K.w + p.w * (1.f - K.w));
        orr.x = (_Float16)(a.x * R.x + p.x * (1.f - R.x));
        orr.y = (_Float16)(a.y * R.y + p.y * (1.f - R.y));
        orr.z = (_Float16)(a.z * R.z + p.z * (1.f - R.z));
        orr.w = (_Float16)(a.w * R.w + p.w * (1.f - R.w));
        ((h4*)xck)[i] = ok;
        ((h4*)xcr)[i] = orr;
    }
}

// ---------------------------------------------------------------------------
// NT GEMM with WMMA f16->f32:  out[m,n] = sum_k A[m,k] * W[n,k]
// Block tile 128x128, 8 waves, wave tile 32x64 (2x4 WMMA accums), K stage 64.
// blockIdx.x = N-block (fastest varying): co-resident blocks share the same
// 256KB activation stripe (L2 hit) while the whole weight matrix (2-8MB)
// stays resident in the 192MB L2 -> activations stream from HBM ~once.
// EPI = 0: store fp32.  EPI = 1: relu()^2 -> f16 (for Wck).
// M,N,K all multiples of 128 here (no bounds checks needed).
// ---------------------------------------------------------------------------
template <int EPI>
__global__ __launch_bounds__(256)
void gemm_nt_wmma(const _Float16* __restrict__ A, const _Float16* __restrict__ W,
                  float* __restrict__ outF, _Float16* __restrict__ outH,
                  int M, int N, int K) {
    constexpr int BM = 128, BN = 128, KT = 64, LDT = KT + 8;  // pad: conflict-free 16B lanes
    __shared__ _Float16 sA[BM * LDT];
    __shared__ _Float16 sB[BN * LDT];

    const int bn = blockIdx.x;   // N-block, fastest varying -> A stripe reuse in L2
    const int bm = blockIdx.y;   // M-block

    const int tid  = threadIdx.x;
    const int lane = tid & 31;
    const int wave = tid >> 5;
    const int wm = (wave & 3) * 32;   // 4 waves along M
    const int wn = (wave >> 2) * 64;  // 2 waves along N

    const int ldr = tid >> 1;                 // 0..127: tile row loaded by this thread
    const int seg = (tid & 1) * 32;           // 0 or 32 halves within K stage
    const size_t rowA = (size_t)(bm * BM + ldr) * K;
    const size_t rowB = (size_t)(bn * BN + ldr) * K;

    const int m16 = lane & 15;
    const int hh  = lane >> 4;

    v8f acc[2][4] = {};

    for (int k0 = 0; k0 < K; k0 += KT) {
        // stage global -> regs
        const float4* gA = (const float4*)(A + rowA + k0 + seg);
        const float4* gB = (const float4*)(W + rowB + k0 + seg);
        float4 a0 = gA[0], a1 = gA[1], a2 = gA[2], a3 = gA[3];
        float4 b0 = gB[0], b1 = gB[1], b2 = gB[2], b3 = gB[3];
        if (k0 + KT < K) {                    // hint next stage into near caches
            __builtin_prefetch(A + rowA + k0 + KT + seg, 0, 3);
            __builtin_prefetch(W + rowB + k0 + KT + seg, 0, 3);
        }
        __syncthreads();                      // previous stage reads done
        float4* pA = (float4*)&sA[ldr * LDT + seg];
        pA[0] = a0; pA[1] = a1; pA[2] = a2; pA[3] = a3;
        float4* pB = (float4*)&sB[ldr * LDT + seg];
        pB[0] = b0; pB[1] = b1; pB[2] = b2; pB[3] = b3;
        __syncthreads();                      // stage visible

#pragma unroll
        for (int ks = 0; ks < KT; ks += 32) {
            // A fragments per ISA 16-bit A 16x32 layout:
            // lane m (lo half): K in [h*8, h*8+8) U [16+h*8, 16+h*8+8)
            v16h af[2];
#pragma unroll
            for (int i = 0; i < 2; i++) {
                const _Float16* p = &sA[(wm + i * 16 + m16) * LDT + ks + hh * 8];
                v8h lo = *(const v8h*)p;
                v8h hi = *(const v8h*)(p + 16);
                v16h f;
#pragma unroll
                for (int q = 0; q < 8; q++) { f[q] = lo[q]; f[q + 8] = hi[q]; }
                af[i] = f;
            }
#pragma unroll
            for (int j = 0; j < 4; j++) {
                // B fragment: lane (n, h) takes K in [h*8, h*8+8) U [16+h*8, 16+h*8+8)
                const _Float16* p = &sB[(wn + j * 16 + m16) * LDT + ks + hh * 8];
                v8h lo = *(const v8h*)p;
                v8h hi = *(const v8h*)(p + 16);
                v16h bf;
#pragma unroll
                for (int q = 0; q < 8; q++) { bf[q] = lo[q]; bf[q + 8] = hi[q]; }
#pragma unroll
                for (int i = 0; i < 2; i++) {
                    acc[i][j] = __builtin_amdgcn_wmma_f32_16x16x32_f16(
                        false, af[i], false, bf, (short)0, acc[i][j], false, false);
                }
            }
        }
    }

    // Epilogue. C/D layout: VGPR r -> M = r + 8*(lane>=16), N = lane&15
    const int cn = m16;
    const int rb = hh * 8;
#pragma unroll
    for (int i = 0; i < 2; i++) {
        const int gmBase = bm * BM + wm + i * 16 + rb;
#pragma unroll
        for (int j = 0; j < 4; j++) {
            const int gn = bn * BN + wn + j * 16 + cn;
#pragma unroll
            for (int r = 0; r < 8; r++) {
                const size_t o = (size_t)(gmBase + r) * N + gn;
                const float val = acc[i][j][r];
                if (EPI == 1) {
                    float h = val > 0.f ? val : 0.f;
                    outH[o] = (_Float16)(h * h);
                } else {
                    outF[o] = val;
                }
            }
        }
    }
}

// ---------------------------------------------------------------------------
// Fused WKV recurrence + x1 = x + sigmoid(r) * wkv
// One thread per (batch, channel); serial over T with log-sum-exp stabilization.
// Launched as 8 blocks x 1024 threads: 8 waves per SIMD so each SIMD
// interleaves 8 independent chains and keeps the transcendental pipe busy.
// ---------------------------------------------------------------------------
__global__ __launch_bounds__(1024)
void wkv_fused(const float* __restrict__ k, const float* __restrict__ v,
               const float* __restrict__ r, const float* __restrict__ w,
               const float* __restrict__ u, const float* __restrict__ x,
               float* __restrict__ x1) {
    const int g = blockIdx.x * blockDim.x + threadIdx.x;  // 0..B*C-1
    const int b = g >> 10;           // / C
    const int c = g & (Cdim - 1);
    const float ww = w[c];
    const float uu = u[c];
    float aa = 0.f, bb = 0.f, pp = -1e38f;
    size_t idx = (size_t)b * Tlen * Cdim + c;
    for (int t = 0; t < Tlen; t++, idx += Cdim) {
        const float kk = k[idx];
        const float vv = v[idx];
        const float p  = fmaxf(pp, uu + kk);
        const float e1 = __expf(pp - p);
        const float e2 = __expf(uu + kk - p);
        const float y  = (e1 * aa + e2 * vv) / (e1 + e2);
        const float p2  = fmaxf(pp + ww, ww + kk);
        const float e1u = __expf(pp + ww - p2);
        const float e2u = __expf(ww + kk - p2);
        aa = e1u * aa + e2u * vv;
        bb = e1u + e2u;
        pp = p2 + __logf(bb);
        const float rr = r[idx];
        x1[idx] = x[idx] + y / (1.f + __expf(-rr));
    }
}

// ---------------------------------------------------------------------------
// out = x1 + sigmoid(rc) * vc
// ---------------------------------------------------------------------------
__global__ __launch_bounds__(256)
void final_mix(const float* __restrict__ x1, const float* __restrict__ rc,
               const float* __restrict__ vc, float* __restrict__ out, int n) {
    int i = blockIdx.x * blockDim.x + threadIdx.x;
    const int stride = gridDim.x * blockDim.x;
    for (; i < n; i += stride) {
        out[i] = x1[i] + vc[i] / (1.f + __expf(-rc[i]));
    }
}

// ---------------------------------------------------------------------------
extern "C" void kernel_launch(void* const* d_in, const int* in_sizes, int n_in,
                              void* d_out, int out_size, void* d_ws, size_t ws_size,
                              hipStream_t stream) {
    const float* x    = (const float*)d_in[0];
    const float* w    = (const float*)d_in[1];
    const float* u    = (const float*)d_in[2];
    const float* rk   = (const float*)d_in[3];
    const float* rv   = (const float*)d_in[4];
    const float* rr   = (const float*)d_in[5];
    const float* crk  = (const float*)d_in[6];
    const float* crr  = (const float*)d_in[7];
    const float* Wk   = (const float*)d_in[8];
    const float* Wv   = (const float*)d_in[9];
    const float* Wr   = (const float*)d_in[10];
    const float* Wck  = (const float*)d_in[11];
    const float* Wcv  = (const float*)d_in[12];
    const float* Wcr  = (const float*)d_in[13];
    const float* ln1g = (const float*)d_in[14];
    const float* ln1b = (const float*)d_in[15];
    const float* ln2g = (const float*)d_in[16];
    const float* ln2b = (const float*)d_in[17];
    float* out = (float*)d_out;

    const size_t F  = (size_t)BT * Cdim;   // 33,554,432 elements
    const size_t FH = (size_t)BT * Hdim;

    char* ws = (char*)d_ws;
    float* xa = (float*)ws;  ws += F * 4;   // LN output (reused for LN2)
    float* kF = (float*)ws;  ws += F * 4;   // k  (later: vc)
    float* vF = (float*)ws;  ws += F * 4;   // v  (later: rc)
    float* rF = (float*)ws;  ws += F * 4;   // r
    float* x1 = (float*)ws;  ws += F * 4;   // residual after time-mix
    _Float16* xk16 = (_Float16*)ws; ws += F * 2;   // later: xck
    _Float16* xv16 = (_Float16*)ws; ws += F * 2;
    _Float16* xr16 = (_Float16*)ws; ws += F * 2;   // later: xcr
    _Float16* kc16 = (_Float16*)ws; ws += FH * 2;  // relu^2 hidden, f16
    _Float16* wk16  = (_Float16*)ws; ws += (size_t)Cdim * Cdim * 2;
    _Float16* wv16  = (_Float16*)ws; ws += (size_t)Cdim * Cdim * 2;
    _Float16* wr16  = (_Float16*)ws; ws += (size_t)Cdim * Cdim * 2;
    _Float16* wcr16 = (_Float16*)ws; ws += (size_t)Cdim * Cdim * 2;
    _Float16* wck16 = (_Float16*)ws; ws += (size_t)Hdim * Cdim * 2;
    _Float16* wcv16 = (_Float16*)ws; ws += (size_t)Cdim * Hdim * 2;

    const int CC = Cdim * Cdim, HC = Hdim * Cdim;
    cvt_f32_f16<<<1024, 256, 0, stream>>>(Wk,  wk16,  CC);
    cvt_f32_f16<<<1024, 256, 0, stream>>>(Wv,  wv16,  CC);
    cvt_f32_f16<<<1024, 256, 0, stream>>>(Wr,  wr16,  CC);
    cvt_f32_f16<<<1024, 256, 0, stream>>>(Wcr, wcr16, CC);
    cvt_f32_f16<<<2048, 256, 0, stream>>>(Wck, wck16, HC);
    cvt_f32_f16<<<2048, 256, 0, stream>>>(Wcv, wcv16, HC);

    // ---- time mixing ----
    ln1024<<<BT, 256, 0, stream>>>(x, ln1g, ln1b, xa);
    mix_tm<<<4096, 256, 0, stream>>>(xa, rk, rv, rr, xk16, xv16, xr16);

    dim3 gKVR(Cdim / 128, BT / 128);   // (N-blocks=8, M-blocks=256)
    gemm_nt_wmma<0><<<gKVR, 256, 0, stream>>>(xk16, wk16, kF, nullptr, BT, Cdim, Cdim);
    gemm_nt_wmma<0><<<gKVR, 256, 0, stream>>>(xv16, wv16, vF, nullptr, BT, Cdim, Cdim);
    gemm_nt_wmma<0><<<gKVR, 256, 0, stream>>>(xr16, wr16, rF, nullptr, BT, Cdim, Cdim);

    wkv_fused<<<(Bsz * Cdim) / 1024, 1024, 0, stream>>>(kF, vF, rF, w, u, x, x1);

    // ---- channel mixing ----
    ln1024<<<BT, 256, 0, stream>>>(x1, ln2g, ln2b, xa);
    mix_cm<<<4096, 256, 0, stream>>>(xa, crk, crr, xk16, xr16);

    dim3 gCK(Hdim / 128, BT / 128);    // (N-blocks=32, M-blocks=256)
    gemm_nt_wmma<1><<<gCK, 256, 0, stream>>>(xk16, wck16, nullptr, kc16, BT, Hdim, Cdim);
    gemm_nt_wmma<0><<<gKVR, 256, 0, stream>>>(kc16, wcv16, kF, nullptr, BT, Cdim, Hdim);  // vc
    gemm_nt_wmma<0><<<gKVR, 256, 0, stream>>>(xr16, wcr16, vF, nullptr, BT, Cdim, Cdim);  // rc

    final_mix<<<4096, 256, 0, stream>>>(x1, vF, kF, out, (int)F);
}
